// RobertaSelfAttention_matchKV_45200235823395
// MI455X (gfx1250) — matrix operand order repaired and checked by reading, not compiled, not run on hardware
//
#include <hip/hip_runtime.h>

typedef __attribute__((ext_vector_type(16))) _Float16 v16h;
typedef __attribute__((ext_vector_type(8)))  _Float16 v8h;
typedef __attribute__((ext_vector_type(8)))  float    v8f;

#define TOKENS 8192
#define NDIM   1024
#define KDIM   1024
#define SEQL   2048
#define NHEAD  16

// LDS B-panel row stride in halves (80B = 20 words): lm*20 mod 64 partitions the
// 64 banks into 16 disjoint 4-bank groups -> conflict-free b128 fragment reads.
#define BSTRIDE 40

__device__ __forceinline__ v16h cat16(v8h lo, v8h hi)
{
    return __builtin_shufflevector(lo, hi, 0,1,2,3,4,5,6,7,8,9,10,11,12,13,14,15);
}

__device__ __forceinline__ v16h ldfrag(const _Float16* bb, int t, int lm, int lhi)
{
    const _Float16* bp = bb + (t * 16 + lm) * BSTRIDE + lhi * 8;
    v8h lo = *(const v8h*)(bp);
    v8h hi = *(const v8h*)(bp + 16);
    return cat16(lo, hi);
}

__device__ __forceinline__ void compute_chunk(const _Float16* bb, v16h a,
                                              v8f acc[8], int lm, int lhi)
{
    v16h b0 = ldfrag(bb, 0, lm, lhi);
    v16h b1 = ldfrag(bb, 1, lm, lhi);
    v16h b2 = ldfrag(bb, 2, lm, lhi);
    v16h b3 = ldfrag(bb, 3, lm, lhi);
    v16h b4 = ldfrag(bb, 4, lm, lhi);
    v16h b5 = ldfrag(bb, 5, lm, lhi);
    v16h b6 = ldfrag(bb, 6, lm, lhi);
    v16h b7 = ldfrag(bb, 7, lm, lhi);
    // Keep all 16 ds_load_b128 in flight before the first WMMA: forces 8 distinct
    // fragment register sets -> partial dscnt waits instead of dscnt==0 per WMMA.
    __builtin_amdgcn_sched_barrier(0);
    acc[0] = __builtin_amdgcn_wmma_f32_16x16x32_f16(false, a, false, b0, (short)0, acc[0], false, false);
    acc[1] = __builtin_amdgcn_wmma_f32_16x16x32_f16(false, a, false, b1, (short)0, acc[1], false, false);
    acc[2] = __builtin_amdgcn_wmma_f32_16x16x32_f16(false, a, false, b2, (short)0, acc[2], false, false);
    acc[3] = __builtin_amdgcn_wmma_f32_16x16x32_f16(false, a, false, b3, (short)0, acc[3], false, false);
    acc[4] = __builtin_amdgcn_wmma_f32_16x16x32_f16(false, a, false, b4, (short)0, acc[4], false, false);
    acc[5] = __builtin_amdgcn_wmma_f32_16x16x32_f16(false, a, false, b5, (short)0, acc[5], false, false);
    acc[6] = __builtin_amdgcn_wmma_f32_16x16x32_f16(false, a, false, b6, (short)0, acc[6], false, false);
    acc[7] = __builtin_amdgcn_wmma_f32_16x16x32_f16(false, a, false, b7, (short)0, acc[7], false, false);
}

// ---------------------------------------------------------------------------
// f32 -> f16 elementwise convert (X)
// ---------------------------------------------------------------------------
__global__ __launch_bounds__(256) void f32_to_f16_kernel(
    const float* __restrict__ in, _Float16* __restrict__ out, int n)
{
    int i = blockIdx.x * 256 + threadIdx.x;
    if (i < n) out[i] = (_Float16)in[i];
}

// ---------------------------------------------------------------------------
// W[k][n] f32 (1024x1024) -> WT[n][k] f16, LDS-tiled transpose
// ---------------------------------------------------------------------------
__global__ __launch_bounds__(256) void transpose_f16_kernel(
    const float* __restrict__ W, _Float16* __restrict__ WT)
{
    __shared__ _Float16 tile[32][33];
    int bx = blockIdx.x * 32;           // n block
    int by = blockIdx.y * 32;           // k block
    int tx = threadIdx.x & 31;
    int ty = threadIdx.x >> 5;          // 0..7
#pragma unroll
    for (int i = 0; i < 32; i += 8)
        tile[ty + i][tx] = (_Float16)W[(size_t)(by + ty + i) * NDIM + bx + tx];
    __syncthreads();
#pragma unroll
    for (int i = 0; i < 32; i += 8)
        WT[(size_t)(bx + ty + i) * KDIM + by + tx] = tile[tx][ty + i];
}

// ---------------------------------------------------------------------------
// GEMM: relu(X @ W + b). 8 waves/block, 128x128 block tile, wave = 16 rows.
// B panel staged through double-buffered LDS (8KB chunks, padded rows).
// Manual 2-phase unroll: no runtime buffer index -> accumulators stay pinned.
// compute_dot==0 : write relu result to Out (V projection, f32)
// compute_dot==1 : fuse dot with ReadingHead, write dotv[token*16+head] only
// ---------------------------------------------------------------------------
__global__ __launch_bounds__(256) void gemm_relu_wmma(
    const _Float16* __restrict__ X, const _Float16* __restrict__ WT,
    const float* __restrict__ bias, float* __restrict__ Out,
    const float* __restrict__ RH, float* __restrict__ dotv, int compute_dot)
{
    __shared__ _Float16 bsh0[128 * BSTRIDE];   // 10 KB
    __shared__ _Float16 bsh1[128 * BSTRIDE];   // 10 KB

    const int lane = threadIdx.x & 31;
    const int wave = threadIdx.x >> 5;
    const int lm   = lane & 15;
    const int lhi  = lane >> 4;
    const int m0   = blockIdx.y * 128 + wave * 16;
    const int n0   = blockIdx.x * 128;

    // B staging mapping: thread -> (row, 16-half segment)
    const int r   = threadIdx.x >> 1;
    const int seg = threadIdx.x & 1;
    const _Float16* gsrc = WT + (size_t)(n0 + r) * KDIM + seg * 16;
    _Float16* sdst0 = &bsh0[r * BSTRIDE + seg * 16];
    _Float16* sdst1 = &bsh1[r * BSTRIDE + seg * 16];

    // A fragment base (16-bit A layout: lanes 0-15 K={0..7,16..23}, 16-31 K={8..15,24..31})
    const _Float16* arow = X + (size_t)(m0 + lm) * KDIM + lhi * 8;

    v8f acc[8] = {};

    // ---- stage chunk 0 -> buf0, load A(0) ----
    {
        v8h lo = *(const v8h*)(gsrc);
        v8h hi = *(const v8h*)(gsrc + 8);
        *(v8h*)(sdst0)     = lo;
        *(v8h*)(sdst0 + 8) = hi;
    }
    v16h a0 = cat16(*(const v8h*)(arow), *(const v8h*)(arow + 16));
    __syncthreads();

    for (int c = 0; c < 30; c += 2) {
        const int k1 = (c + 1) * 32;
        const int k2 = (c + 2) * 32;

        // phase 0: prefetch chunk c+1, compute chunk c from buf0
        v8h g1lo = *(const v8h*)(gsrc + k1);
        v8h g1hi = *(const v8h*)(gsrc + k1 + 8);
        v16h a1 = cat16(*(const v8h*)(arow + k1), *(const v8h*)(arow + k1 + 16));
        compute_chunk(bsh0, a0, acc, lm, lhi);
        __syncthreads();
        *(v8h*)(sdst1)     = g1lo;
        *(v8h*)(sdst1 + 8) = g1hi;
        __syncthreads();

        // phase 1: prefetch chunk c+2, compute chunk c+1 from buf1
        v8h g2lo = *(const v8h*)(gsrc + k2);
        v8h g2hi = *(const v8h*)(gsrc + k2 + 8);
        a0 = cat16(*(const v8h*)(arow + k2), *(const v8h*)(arow + k2 + 16));
        compute_chunk(bsh1, a1, acc, lm, lhi);
        __syncthreads();
        *(v8h*)(sdst0)     = g2lo;
        *(v8h*)(sdst0 + 8) = g2hi;
        __syncthreads();
    }

    // ---- chunks 30 (buf0) and 31 (buf1) ----
    {
        const int k1 = 31 * 32;
        v8h g1lo = *(const v8h*)(gsrc + k1);
        v8h g1hi = *(const v8h*)(gsrc + k1 + 8);
        v16h a1 = cat16(*(const v8h*)(arow + k1), *(const v8h*)(arow + k1 + 16));
        compute_chunk(bsh0, a0, acc, lm, lhi);
        __syncthreads();
        *(v8h*)(sdst1)     = g1lo;
        *(v8h*)(sdst1 + 8) = g1hi;
        __syncthreads();
        compute_chunk(bsh1, a1, acc, lm, lhi);
    }

    if (!compute_dot) {
#pragma unroll
        for (int t = 0; t < 8; ++t) {
            int n = n0 + t * 16 + lm;
            float bv = bias[n];
#pragma unroll
            for (int i = 0; i < 8; ++i) {
                int m = m0 + lhi * 8 + i;
                float v = acc[t][i] + bv;
                Out[(size_t)m * NDIM + n] = v > 0.f ? v : 0.f;
            }
        }
    } else {
        // dot[m][head] = sum_d relu(K1[m][head*64+d]) * RH[head*64+d]
        // block covers heads n0/64 .. n0/64+1 fully (no atomics needed)
#pragma unroll
        for (int hh = 0; hh < 2; ++hh) {
            float dsum[8];
#pragma unroll
            for (int i = 0; i < 8; ++i) dsum[i] = 0.f;
#pragma unroll
            for (int tt = 0; tt < 4; ++tt) {
                int t = hh * 4 + tt;
                int n = n0 + t * 16 + lm;
                float bv = bias[n];
                float rv = RH[n];           // flat index h*64+d == n
#pragma unroll
                for (int i = 0; i < 8; ++i) {
                    float v = acc[t][i] + bv;
                    v = v > 0.f ? v : 0.f;
                    dsum[i] += v * rv;
                }
            }
#pragma unroll
            for (int i = 0; i < 8; ++i) {
                float s = dsum[i];
                s += __shfl_xor(s, 1);
                s += __shfl_xor(s, 2);
                s += __shfl_xor(s, 4);
                s += __shfl_xor(s, 8);
                if (lm == 0) {
                    int m = m0 + lhi * 8 + i;
                    int head = (n0 >> 6) + hh;
                    dotv[(size_t)m * NHEAD + head] = s;
                }
            }
        }
    }
}

// ---------------------------------------------------------------------------
// Per-(b,h) parallel scan: valid = dot > 0.5 (pos 0 masked).
// P = compacted valid positions in [1, L-1] (increasing).
// forward regs at l: P[C(l)-1-r] (0 if short)    [C(l)=#valid in 1..l]
// backward regs at l (1..L-2): r<S -> P[C(l-1)+r]+1 ; r==S -> last ; else 0
//   with S = countq - C(l-1), countq = |P| - valid(L-1), last = valid(L-1)?L-1:0
// l==0 -> zeros ; l==L-1 -> [last,0,0,0]
// idx8 layout: [token][head][8] (0..3 fwd, 4..7 bwd)
// ---------------------------------------------------------------------------
__global__ __launch_bounds__(256) void scan_kernel(
    const float* __restrict__ dotv, int* __restrict__ idx8)
{
    const int b = blockIdx.x >> 4;
    const int h = blockIdx.x & 15;
    __shared__ unsigned short P[SEQL];
    __shared__ int warp_base[8];
    __shared__ int sh_total;
    __shared__ int sh_lastvalid;

    int t = threadIdx.x;
    int lane = t & 31, w = t >> 5;
    int base_l = t * 8;

    unsigned flags = 0;
#pragma unroll
    for (int j = 0; j < 8; ++j) {
        int l = base_l + j;
        float dv = dotv[((size_t)(b * SEQL + l) << 4) + h];
        if (dv > 0.5f && l != 0) flags |= (1u << j);
    }
    int cnt = __popc(flags);
    int incl = cnt;
#pragma unroll
    for (int off = 1; off < 32; off <<= 1) {
        int v = __shfl_up(incl, off);
        if (lane >= off) incl += v;
    }
    if (lane == 31) warp_base[w] = incl;
    if (t == 255) sh_lastvalid = (int)((flags >> 7) & 1u);
    __syncthreads();
    if (t == 0) {
        int s = 0;
        for (int i = 0; i < 8; ++i) { int v = warp_base[i]; warp_base[i] = s; s += v; }
        sh_total = s;
    }
    __syncthreads();
    int excl = warp_base[w] + (incl - cnt);
    int c = excl;
#pragma unroll
    for (int j = 0; j < 8; ++j)
        if ((flags >> j) & 1u) P[c++] = (unsigned short)(base_l + j);
    __syncthreads();

    int total  = sh_total;
    int lastv  = sh_lastvalid;
    int last   = lastv ? (SEQL - 1) : 0;
    int countq = total - lastv;

    int C = excl;
    int* outp = idx8 + ((size_t)((b * SEQL + base_l) * NHEAD + h) << 3);
#pragma unroll
    for (int j = 0; j < 8; ++j) {
        int l = base_l + j;
        int Cprev = C;
        if ((flags >> j) & 1u) C++;
        int Cl = C;
        int r0 = (Cl >= 1) ? (int)P[Cl - 1] : 0;
        int r1 = (Cl >= 2) ? (int)P[Cl - 2] : 0;
        int r2 = (Cl >= 3) ? (int)P[Cl - 3] : 0;
        int r3 = (Cl >= 4) ? (int)P[Cl - 4] : 0;
        int s0, s1, s2, s3;
        if (l == 0) {
            s0 = s1 = s2 = s3 = 0;
        } else if (l == SEQL - 1) {
            s0 = last; s1 = s2 = s3 = 0;
        } else {
            int S = countq - Cprev;
            s0 = (0 < S) ? (int)P[Cprev + 0] + 1 : ((0 == S) ? last : 0);
            s1 = (1 < S) ? (int)P[Cprev + 1] + 1 : ((1 == S) ? last : 0);
            s2 = (2 < S) ? (int)P[Cprev + 2] + 1 : ((2 == S) ? last : 0);
            s3 = (3 < S) ? (int)P[Cprev + 3] + 1 : ((3 == S) ? last : 0);
        }
        int* o = outp + (size_t)j * NHEAD * 8;
        ((int4*)o)[0] = make_int4(r0, r1, r2, r3);
        ((int4*)o)[1] = make_int4(s0, s1, s2, s3);
    }
}

// ---------------------------------------------------------------------------
// Gather + weighted sum: out[b,l,h,d] = sum_r w[h][r] * V1[b, idx_r, h, d]
// ---------------------------------------------------------------------------
__global__ __launch_bounds__(256) void gather_kernel(
    const float* __restrict__ V1, const int* __restrict__ idx8,
    const float* __restrict__ bw, float* __restrict__ out)
{
    int g    = blockIdx.x * 256 + threadIdx.x;    // ((bl*16)+h)*64 + d
    int d    = g & 63;
    int rest = g >> 6;
    int h    = rest & 15;
    int bl   = rest >> 4;                         // b*2048 + l
    int b    = bl >> 11;
    const int*   ix = idx8 + ((size_t)(bl * NHEAD + h) << 3);
    const float* w  = bw + h * 8;                 // [0..3]=fwd, [4..7]=bwd
    float acc = 0.f;
#pragma unroll
    for (int r = 0; r < 8; ++r) {
        int pos = ix[r];
        acc += w[r] * V1[(size_t)((b << 11) + pos) * NDIM + h * 64 + d];
    }
    out[(size_t)g] = acc;
}

// ---------------------------------------------------------------------------
extern "C" void kernel_launch(void* const* d_in, const int* in_sizes, int n_in,
                              void* d_out, int out_size, void* d_ws, size_t ws_size,
                              hipStream_t stream)
{
    (void)in_sizes; (void)n_in; (void)out_size; (void)ws_size;
    const float* hs = (const float*)d_in[0];
    const float* Wk = (const float*)d_in[1];
    const float* bk = (const float*)d_in[2];
    const float* Wv = (const float*)d_in[3];
    const float* bv = (const float*)d_in[4];
    const float* RH = (const float*)d_in[5];
    const float* bw = (const float*)d_in[6];
    float* out = (float*)d_out;

    char* ws = (char*)d_ws;
    _Float16* Xh   = (_Float16*)(ws);                        // 16 MB
    _Float16* WkT  = (_Float16*)(ws + (size_t)(16u << 20));  //  2 MB
    _Float16* WvT  = (_Float16*)(ws + (size_t)(18u << 20));  //  2 MB
    float*    V1   = (float*)   (ws + (size_t)(20u << 20));  // 32 MB
    float*    dotv = (float*)   (ws + (size_t)(52u << 20));  // 0.5 MB
    int*      idx8 = (int*)     (ws + (size_t)(53u << 20));  //  4 MB

    f32_to_f16_kernel<<<(TOKENS * KDIM) / 256, 256, 0, stream>>>(hs, Xh, TOKENS * KDIM);
    transpose_f16_kernel<<<dim3(32, 32), 256, 0, stream>>>(Wk, WkT);
    transpose_f16_kernel<<<dim3(32, 32), 256, 0, stream>>>(Wv, WvT);

    // V projection: store relu result
    gemm_relu_wmma<<<dim3(8, 64), 256, 0, stream>>>(Xh, WvT, bv, V1, RH, dotv, 0);
    // K projection: fused dot with ReadingHead, K1 never materialized
    gemm_relu_wmma<<<dim3(8, 64), 256, 0, stream>>>(Xh, WkT, bk, (float*)nullptr, RH, dotv, 1);

    scan_kernel<<<64, 256, 0, stream>>>(dotv, idx8);
    gather_kernel<<<(TOKENS * NHEAD * 64) / 256, 256, 0, stream>>>(V1, idx8, bw, out);
}
